// _LBlockSpatial_5961414607317
// MI455X (gfx1250) — matrix operand order repaired and checked by reading, not compiled, not run on hardware
//
#include <hip/hip_runtime.h>
#include <hip/hip_bf16.h>

#define N_NODES 50000
#define DIM 128
#define NT 4
#define NE 150000
#define NH 8

typedef _Float16 half8  __attribute__((ext_vector_type(8)));
typedef _Float16 half16 __attribute__((ext_vector_type(16)));
typedef float    float8 __attribute__((ext_vector_type(8)));
typedef int      i4v    __attribute__((ext_vector_type(4)));
typedef __attribute__((address_space(1))) i4v* as1_i4v;
typedef __attribute__((address_space(3))) i4v* as3_i4v;

struct EIPtrs { const int* p[4]; };

__device__ __forceinline__ float geluf(float x) {
  return 0.5f * x * (1.0f + erff(x * 0.7071067811865476f));
}
// order-preserving float<->uint encoding for atomic max over floats
__device__ __forceinline__ unsigned fenc(float f) {
  unsigned b = __float_as_uint(f);
  return (b & 0x80000000u) ? ~b : (b | 0x80000000u);
}
__device__ __forceinline__ float fdec(unsigned u) {
  unsigned b = (u & 0x80000000u) ? (u & 0x7FFFFFFFu) : ~u;
  return __uint_as_float(b);
}

// ---------------------------------------------------------------------------
// prep: transpose+f16 all weight matrices; build block-diag arel/mrel mats;
// pack biases. wf16 slots: 0-3 W1[t]^T, 4-7 W2[t]^T, 8 Wk^T, 9 Wq^T, 10 Wv^T,
// 11 Wo^T, 12-15 blkdiag(arel[t])^T, 16-19 blkdiag(mrel[t])^T.
// bpack rows: 0 bk, 1 bq, 2 bv, 3 bo.  bsum rows: 0 sum_t b1, 1 sum_t b2.
// bzero: 128 zeros (branchless "no bias" row).
// ---------------------------------------------------------------------------
__global__ __launch_bounds__(256) void prep_kernel(
    const float* __restrict__ W1, const float* __restrict__ W2,
    const float* __restrict__ Wk, const float* __restrict__ Wq,
    const float* __restrict__ Wv, const float* __restrict__ Wo,
    const float* __restrict__ arel, const float* __restrict__ mrel,
    const float* __restrict__ bk, const float* __restrict__ bq,
    const float* __restrict__ bv, const float* __restrict__ bo,
    const float* __restrict__ b1, const float* __restrict__ b2,
    _Float16* __restrict__ wf16, float* __restrict__ bpack,
    float* __restrict__ bsum, float* __restrict__ bzero)
{
  const int s = blockIdx.x;
  const int tid = threadIdx.x;
  if (s < 20) {
    for (int i = tid; i < 16384; i += 256) {
      int c = i >> 7, k = i & 127;  // store WT[c][k] = W[k][c]
      float v;
      if (s < 4)        v = W1[(size_t)s * 16384 + k * 128 + c];
      else if (s < 8)   v = W2[(size_t)(s - 4) * 16384 + k * 128 + c];
      else if (s == 8)  v = Wk[k * 128 + c];
      else if (s == 9)  v = Wq[k * 128 + c];
      else if (s == 10) v = Wv[k * 128 + c];
      else if (s == 11) v = Wo[k * 128 + c];
      else if (s < 16) { // blkdiag(arel[t]): W[h*16+f][h*16+d] = arel[t][h][d][f]
        int t = s - 12; int hc = c >> 4, hk = k >> 4;
        v = (hc == hk) ? arel[(((size_t)t * NH + hc) * 16 + (c & 15)) * 16 + (k & 15)] : 0.0f;
      } else {
        int t = s - 16; int hc = c >> 4, hk = k >> 4;
        v = (hc == hk) ? mrel[(((size_t)t * NH + hc) * 16 + (c & 15)) * 16 + (k & 15)] : 0.0f;
      }
      wf16[(size_t)s * 16384 + i] = (_Float16)v;
    }
  } else if (tid < 128) {
    bpack[tid] = bk[tid]; bpack[128 + tid] = bq[tid];
    bpack[256 + tid] = bv[tid]; bpack[384 + tid] = bo[tid];
    float s1 = 0.f, s2 = 0.f;
    for (int t = 0; t < NT; ++t) { s1 += b1[t * 128 + tid]; s2 += b2[t * 128 + tid]; }
    bsum[tid] = s1; bsum[128 + tid] = s2;
    bzero[tid] = 0.0f;
  }
}

// ---------------------------------------------------------------------------
// WMMA GEMM: Out[mat][r][c] = sum_k A[r][k]*W[k][c] + bias[mat*biasStride+c],
// with WT[c][k] (f16) per mat. Block = 8 waves x 16 rows = 128 rows.
// Weight tile staged to LDS via async-to-LDS copy when available.
// ---------------------------------------------------------------------------
template <bool OUT_F16>
__global__ __launch_bounds__(256) void gemm128(
    const _Float16* __restrict__ A, const _Float16* __restrict__ WT,
    void* __restrict__ outBase, const float* __restrict__ bias,
    long biasStride, int nrows, long outStrideElems)
{
  __shared__ _Float16 sWT[128 * 128];
  const int mat = blockIdx.y;

  const int wave = threadIdx.x >> 5;
  const int lane = threadIdx.x & 31;
  const int ln15 = lane & 15;
  const int hi = lane >> 4;
  const int m0 = blockIdx.x * 128 + wave * 16;

  int rowA = m0 + ln15;
  if (rowA >= nrows) rowA = nrows - 1;   // clamp; stores are guarded
  const _Float16* arow = A + (size_t)rowA * DIM;
  // pull the A row (256B = 2 cachelines) toward the WGP while we stage weights
  __builtin_prefetch(arow, 0, 3);
  __builtin_prefetch(arow + 64, 0, 3);

  {
    uint4* s4 = (uint4*)(WT + (size_t)mat * 16384);
    uint4* d4 = (uint4*)sWT;
#if __has_builtin(__builtin_amdgcn_global_load_async_to_lds_b128)
    for (int i = threadIdx.x; i < 2048; i += 256)
      __builtin_amdgcn_global_load_async_to_lds_b128(
          (as1_i4v)(s4 + i), (as3_i4v)(d4 + i), 0, 0);
#if __has_builtin(__builtin_amdgcn_s_wait_asynccnt)
    __builtin_amdgcn_s_wait_asynccnt(0);
#else
    asm volatile("s_wait_asynccnt 0x0" ::: "memory");
#endif
#else
    for (int i = threadIdx.x; i < 2048; i += 256) d4[i] = s4[i];
#endif
  }
  __syncthreads();

  // A fragments: 16x32 f16 layout -> lanes 0-15: K 0..7 & 16..23; lanes 16-31: +8
  half16 afrag[4];
#pragma unroll
  for (int kc = 0; kc < 4; ++kc) {
    const int kb = kc * 32 + hi * 8;
    half8 lo = *(const half8*)(arow + kb);
    half8 hi8 = *(const half8*)(arow + kb + 16);
    afrag[kc] = __builtin_shufflevector(lo, hi8,
        0, 1, 2, 3, 4, 5, 6, 7, 8, 9, 10, 11, 12, 13, 14, 15);
  }

  const bool fullTile = (m0 + 16) <= nrows;   // wave-uniform

#pragma unroll
  for (int ct = 0; ct < 8; ++ct) {
    const int col = ct * 16 + ln15;
    float8 acc = {};
#pragma unroll
    for (int kc = 0; kc < 4; ++kc) {
      // B fragment: 32x16 layout -> lanes 0-15: K 0..7 & 16..23; lanes 16-31: +8
      const int kb = kc * 32 + hi * 8;
      half8 b0 = *(const half8*)(&sWT[col * 128 + kb]);
      half8 b1 = *(const half8*)(&sWT[col * 128 + kb + 16]);
      half16 bf = __builtin_shufflevector(b0, b1,
          0, 1, 2, 3, 4, 5, 6, 7, 8, 9, 10, 11, 12, 13, 14, 15);
      acc = __builtin_amdgcn_wmma_f32_16x16x32_f16(
          false, afrag[kc], false, bf, (short)0, acc, false, false);
    }
    const float bv = bias[(size_t)mat * biasStride + col];
    if (fullTile) {
#pragma unroll
      for (int j = 0; j < 8; ++j) {
        const int r = m0 + hi * 8 + j;   // C/D layout: row = 8*hi + j, col = ln15
        const float val = acc[j] + bv;
        if (OUT_F16)
          ((_Float16*)outBase)[(size_t)mat * outStrideElems + (size_t)r * DIM + col] =
              (_Float16)val;
        else
          ((float*)outBase)[(size_t)mat * outStrideElems + (size_t)r * DIM + col] = val;
      }
    } else {
#pragma unroll
      for (int j = 0; j < 8; ++j) {
        const int r = m0 + hi * 8 + j;
        if (r < nrows) {
          const float val = acc[j] + bv;
          if (OUT_F16)
            ((_Float16*)outBase)[(size_t)mat * outStrideElems + (size_t)r * DIM + col] =
                (_Float16)val;
          else
            ((float*)outBase)[(size_t)mat * outStrideElems + (size_t)r * DIM + col] = val;
        }
      }
    }
  }
}

// ---------------------------------------------------------------------------
__global__ __launch_bounds__(256) void add3_kernel(
    const float* __restrict__ a, const float* __restrict__ b,
    const float* __restrict__ c, float* __restrict__ h,
    _Float16* __restrict__ h16, long n)
{
  long i = (long)blockIdx.x * blockDim.x + threadIdx.x;
  if (i >= n) return;
  float v = a[i] + b[i] + c[i];
  h[i] = v; h16[i] = (_Float16)v;
}

__global__ __launch_bounds__(256) void fill_u32(unsigned* __restrict__ p, unsigned v, long n)
{
  long i = (long)blockIdx.x * blockDim.x + threadIdx.x;
  if (i < n) p[i] = v;
}

// al/ar: per (t,n,h) 16-dim dot of xw row-slice with attention vectors
__global__ __launch_bounds__(256) void gat_coeff(
    const float* __restrict__ xw, const float* __restrict__ asrc,
    const float* __restrict__ adst, float* __restrict__ al, float* __restrict__ ar)
{
  long gid = (long)blockIdx.x * blockDim.x + threadIdx.x;
  if (gid >= (long)NT * N_NODES * NH) return;
  int h = (int)(gid & 7); long tn = gid >> 3;
  int t = (int)(tn / N_NODES);
  const float* x = xw + tn * DIM + h * 16;
  const float* a1 = asrc + ((size_t)t * NH + h) * 16;
  const float* a2 = adst + ((size_t)t * NH + h) * 16;
  float s1 = 0.f, s2 = 0.f;
#pragma unroll
  for (int d = 0; d < 16; ++d) { float xv = x[d]; s1 += xv * a1[d]; s2 += xv * a2[d]; }
  al[gid] = s1; ar[gid] = s2;
}

// GAT pass A: segment max (per type, per dst, per head), self-loops appended
__global__ __launch_bounds__(256) void gat_edge_max(
    const float* __restrict__ al, const float* __restrict__ ar,
    EIPtrs ei, unsigned* __restrict__ mbuf)
{
  long gid = (long)blockIdx.x * blockDim.x + threadIdx.x;
  const long EPT = (long)NE + N_NODES;
  if (gid >= (long)NT * EPT) return;
  int t = (int)(gid / EPT); long idx = gid - (long)t * EPT;
  int src, dst;
  if (idx < NE) { src = ei.p[t][idx]; dst = ei.p[t][NE + idx]; }
  else { src = dst = (int)(idx - NE); }
  const float* alp = al + ((size_t)t * N_NODES + src) * NH;
  const float* arp = ar + ((size_t)t * N_NODES + dst) * NH;
  unsigned* mp = mbuf + ((size_t)t * N_NODES + dst) * NH;
#pragma unroll
  for (int h = 0; h < NH; ++h) {
    float e = alp[h] + arp[h];
    e = e > 0.f ? e : 0.2f * e;
    atomicMax(&mp[h], fenc(e));
  }
}

// GAT pass B: s[dst,h] += exp(e - m)
__global__ __launch_bounds__(256) void gat_edge_sum(
    const float* __restrict__ al, const float* __restrict__ ar,
    EIPtrs ei, const unsigned* __restrict__ mbuf, float* __restrict__ sbuf)
{
  long gid = (long)blockIdx.x * blockDim.x + threadIdx.x;
  const long EPT = (long)NE + N_NODES;
  if (gid >= (long)NT * EPT) return;
  int t = (int)(gid / EPT); long idx = gid - (long)t * EPT;
  int src, dst;
  if (idx < NE) { src = ei.p[t][idx]; dst = ei.p[t][NE + idx]; }
  else { src = dst = (int)(idx - NE); }
  const float* alp = al + ((size_t)t * N_NODES + src) * NH;
  const float* arp = ar + ((size_t)t * N_NODES + dst) * NH;
  const unsigned* mp = mbuf + ((size_t)t * N_NODES + dst) * NH;
  float* sp = sbuf + ((size_t)t * N_NODES + dst) * NH;
#pragma unroll
  for (int h = 0; h < NH; ++h) {
    float e = alp[h] + arp[h];
    e = e > 0.f ? e : 0.2f * e;
    atomicAdd(&sp[h], expf(e - fdec(mp[h])));
  }
}

// GAT pass C: wave per edge, agg[dst] += xw[src]*alpha (summed across types)
__global__ __launch_bounds__(256) void gat_edge_agg(
    const float* __restrict__ xw, const float* __restrict__ al,
    const float* __restrict__ ar, EIPtrs ei,
    const unsigned* __restrict__ mbuf, const float* __restrict__ sbuf,
    float* __restrict__ agg)
{
  long tidg = (long)blockIdx.x * blockDim.x + threadIdx.x;
  long wid = tidg >> 5; int lane = (int)(tidg & 31);
  const long EPT = (long)NE + N_NODES;
  if (wid >= (long)NT * EPT) return;
  int t = (int)(wid / EPT); long idx = wid - (long)t * EPT;
  int src, dst;
  if (idx < NE) { src = ei.p[t][idx]; dst = ei.p[t][NE + idx]; }
  else { src = dst = (int)(idx - NE); }
  int h = lane >> 2;
  size_t tnd = (size_t)t * N_NODES + dst;
  size_t tns = (size_t)t * N_NODES + src;
  float e = al[tns * NH + h] + ar[tnd * NH + h];
  e = e > 0.f ? e : 0.2f * e;
  float alpha = expf(e - fdec(mbuf[tnd * NH + h])) / sbuf[tnd * NH + h];
  const float4 xv = *(const float4*)(xw + tns * DIM + lane * 4);
  float* ap = agg + (size_t)dst * DIM + lane * 4;
  atomicAdd(ap + 0, xv.x * alpha);
  atomicAdd(ap + 1, xv.y * alpha);
  atomicAdd(ap + 2, xv.z * alpha);
  atomicAdd(ap + 3, xv.w * alpha);
}

// h = gelu(h + rmsnorm(agg + bsum, g)); refresh f16 copy. Wave per row.
__global__ __launch_bounds__(256) void post_norm_gat(
    const float* __restrict__ agg, const float* __restrict__ bsum,
    const float* __restrict__ g, float* __restrict__ h, _Float16* __restrict__ h16)
{
  long tid = (long)blockIdx.x * blockDim.x + threadIdx.x;
  long row = tid >> 5; int lane = (int)(tid & 31);
  if (row >= N_NODES) return;
  float4 a = ((const float4*)(agg + row * DIM))[lane];
  float4 bs = ((const float4*)bsum)[lane];
  float x0 = a.x + bs.x, x1 = a.y + bs.y, x2 = a.z + bs.z, x3 = a.w + bs.w;
  float ss = x0 * x0 + x1 * x1 + x2 * x2 + x3 * x3;
  for (int o = 16; o; o >>= 1) ss += __shfl_xor(ss, o, 32);
  float r = rsqrtf(ss * (1.0f / DIM) + 1e-6f);
  float4 gv = ((const float4*)g)[lane];
  float4 hv = ((const float4*)(h + row * DIM))[lane];
  float y0 = geluf(hv.x + gv.x * x0 * r);
  float y1 = geluf(hv.y + gv.y * x1 * r);
  float y2 = geluf(hv.z + gv.z * x2 * r);
  float y3 = geluf(hv.w + gv.w * x3 * r);
  ((float4*)(h + row * DIM))[lane] = float4{y0, y1, y2, y3};
  _Float16* hp = h16 + row * DIM + lane * 4;
  hp[0] = (_Float16)y0; hp[1] = (_Float16)y1; hp[2] = (_Float16)y2; hp[3] = (_Float16)y3;
}

// HGT attention logits: att[t,e,h] = <k16[src], qa16[t][dst]>_h * prel * 0.25
__global__ __launch_bounds__(256) void hgt_att_kernel(
    const _Float16* __restrict__ k16, const _Float16* __restrict__ qa16,
    const float* __restrict__ prel, EIPtrs ei,
    float* __restrict__ att, unsigned* __restrict__ mbuf)
{
  long gid = (long)blockIdx.x * blockDim.x + threadIdx.x;
  if (gid >= (long)NT * NE * NH) return;
  int h = (int)(gid & 7); long te = gid >> 3;
  int t = (int)(te / NE); long e = te - (long)t * NE;
  int src = ei.p[t][e], dst = ei.p[t][NE + e];
  const _Float16* kp = k16 + (size_t)src * DIM + h * 16;
  const _Float16* qp = qa16 + ((size_t)t * N_NODES + dst) * DIM + h * 16;
  float s = 0.f;
#pragma unroll
  for (int d = 0; d < 16; ++d) s += (float)kp[d] * (float)qp[d];
  s *= prel[t * NH + h] * 0.25f;
  att[gid] = s;
  atomicMax(&mbuf[(size_t)dst * NH + h], fenc(s));
}

__global__ __launch_bounds__(256) void hgt_sum_kernel(
    const float* __restrict__ att, EIPtrs ei,
    const unsigned* __restrict__ mbuf, float* __restrict__ sbuf)
{
  long gid = (long)blockIdx.x * blockDim.x + threadIdx.x;
  if (gid >= (long)NT * NE * NH) return;
  int h = (int)(gid & 7); long te = gid >> 3;
  int t = (int)(te / NE); long e = te - (long)t * NE;
  int dst = ei.p[t][NE + e];
  float p = expf(att[gid] - fdec(mbuf[(size_t)dst * NH + h]));
  atomicAdd(&sbuf[(size_t)dst * NH + h], p);
}

// wave per (t,e): agg[dst] += vm16[t][src]*alpha
__global__ __launch_bounds__(256) void hgt_agg_kernel(
    const float* __restrict__ att, const _Float16* __restrict__ vm16,
    EIPtrs ei, const unsigned* __restrict__ mbuf, const float* __restrict__ sbuf,
    float* __restrict__ agg)
{
  long tidg = (long)blockIdx.x * blockDim.x + threadIdx.x;
  long wid = tidg >> 5; int lane = (int)(tidg & 31);
  if (wid >= (long)NT * NE) return;
  int t = (int)(wid / NE); long e = wid - (long)t * NE;
  int src = ei.p[t][e], dst = ei.p[t][NE + e];
  int h = lane >> 2;
  float alpha = expf(att[wid * NH + h] - fdec(mbuf[(size_t)dst * NH + h])) /
                sbuf[(size_t)dst * NH + h];
  const _Float16* vp = vm16 + ((size_t)t * N_NODES + src) * DIM + lane * 4;
  float* ap = agg + (size_t)dst * DIM + lane * 4;
  atomicAdd(ap + 0, (float)vp[0] * alpha);
  atomicAdd(ap + 1, (float)vp[1] * alpha);
  atomicAdd(ap + 2, (float)vp[2] * alpha);
  atomicAdd(ap + 3, (float)vp[3] * alpha);
}

__global__ __launch_bounds__(256) void gelu16_kernel(
    const float* __restrict__ agg, _Float16* __restrict__ ge16, long n)
{
  long i = (long)blockIdx.x * blockDim.x + threadIdx.x;
  if (i >= n) return;
  ge16[i] = (_Float16)geluf(agg[i]);
}

// out = gelu(h + rmsnorm(beta*O + (1-beta)*h, g3))   (O already has bo)
__global__ __launch_bounds__(256) void final_kernel(
    const float* __restrict__ O, const float* __restrict__ h,
    const float* __restrict__ g3, const float* __restrict__ skip,
    float* __restrict__ out)
{
  long tid = (long)blockIdx.x * blockDim.x + threadIdx.x;
  long row = tid >> 5; int lane = (int)(tid & 31);
  if (row >= N_NODES) return;
  float beta = 1.0f / (1.0f + expf(-skip[0]));
  float4 o = ((const float4*)(O + row * DIM))[lane];
  float4 hv = ((const float4*)(h + row * DIM))[lane];
  float x0 = beta * o.x + (1.f - beta) * hv.x;
  float x1 = beta * o.y + (1.f - beta) * hv.y;
  float x2 = beta * o.z + (1.f - beta) * hv.z;
  float x3 = beta * o.w + (1.f - beta) * hv.w;
  float ss = x0 * x0 + x1 * x1 + x2 * x2 + x3 * x3;
  for (int oo = 16; oo; oo >>= 1) ss += __shfl_xor(ss, oo, 32);
  float r = rsqrtf(ss * (1.0f / DIM) + 1e-6f);
  float4 gv = ((const float4*)g3)[lane];
  ((float4*)(out + row * DIM))[lane] = float4{
      geluf(hv.x + gv.x * x0 * r), geluf(hv.y + gv.y * x1 * r),
      geluf(hv.z + gv.z * x2 * r), geluf(hv.w + gv.w * x3 * r)};
}

// ---------------------------------------------------------------------------
extern "C" void kernel_launch(void* const* d_in, const int* in_sizes, int n_in,
                              void* d_out, int out_size, void* d_ws, size_t ws_size,
                              hipStream_t stream)
{
  (void)in_sizes; (void)n_in; (void)out_size; (void)ws_size;
  const float* zL   = (const float*)d_in[0];
  const float* zH   = (const float*)d_in[1];
  const float* xemb = (const float*)d_in[2];
  const float* W1   = (const float*)d_in[3];
  const float* as1  = (const float*)d_in[4];
  const float* ad1  = (const float*)d_in[5];
  const float* b1   = (const float*)d_in[6];
  const float* W2   = (const float*)d_in[7];
  const float* as2  = (const float*)d_in[8];
  const float* ad2  = (const float*)d_in[9];
  const float* b2   = (const float*)d_in[10];
  const float* Wk   = (const float*)d_in[11];
  const float* bk   = (const float*)d_in[12];
  const float* Wq   = (const float*)d_in[13];
  const float* bq   = (const float*)d_in[14];
  const float* Wv   = (const float*)d_in[15];
  const float* bv   = (const float*)d_in[16];
  const float* arel = (const float*)d_in[17];
  const float* mrel = (const float*)d_in[18];
  const float* prel = (const float*)d_in[19];
  const float* Wo   = (const float*)d_in[20];
  const float* bo   = (const float*)d_in[21];
  const float* skip = (const float*)d_in[22];
  const float* g1   = (const float*)d_in[23];
  const float* g2   = (const float*)d_in[24];
  const float* g3   = (const float*)d_in[25];

  EIPtrs eip;
  for (int t = 0; t < NT; ++t) eip.p[t] = (const int*)d_in[26 + t];

  char* ws = (char*)d_ws;
  const size_t NB = (size_t)N_NODES * DIM;            // 6.4e6 elements
  float*    h    = (float*)(ws + 0);                  // 25.6 MB
  _Float16* h16  = (_Float16*)(ws + 25600000);        // 12.8 MB
  unsigned* mbuf = (unsigned*)(ws + 38400000);        // 6.4 MB  [T][N][H]
  float*    sbuf = (float*)(ws + 44800000);           // 6.4 MB
  float*    agg  = (float*)(ws + 51200000);           // 25.6 MB
  _Float16* wf16 = (_Float16*)(ws + 76800000);        // 20*32KB
  float*    bpack= (float*)(ws + 77455360);           // 4*128 f32
  float*    bsum = (float*)(ws + 77457408);           // 2*128 f32
  float*    bzero= (float*)(ws + 77458432);           // 128 f32 zeros
  char* scr = ws + 77458944;
  // GAT overlay
  float* xw = (float*)scr;                            // 4*N*128 f32 = 102.4 MB
  float* al = (float*)(scr + 102400000);              // 6.4 MB
  float* ar = (float*)(scr + 108800000);              // 6.4 MB
  // HGT overlay (reuses scr)
  _Float16* kqv16 = (_Float16*)scr;                   // 3*N*128 f16 = 38.4 MB
  _Float16* qa16  = (_Float16*)(scr + 38400000);      // 4*N*128 f16 = 51.2 MB
  _Float16* vm16  = (_Float16*)(scr + 89600000);      // 51.2 MB
  float*    att   = (float*)(scr + 140800000);        // 4*E*8 f32 = 19.2 MB
  _Float16* ge16  = (_Float16*)(scr + 160000000);     // 12.8 MB
  float*    Obuf  = (float*)(scr + 172800000);        // 25.6 MB

  const long EPT = (long)NE + N_NODES;                // 200000 (with self loops)
  const int gemmGx = (N_NODES + 127) / 128;           // 391
  const long nbThreads = (long)NB;

  prep_kernel<<<21, 256, 0, stream>>>(W1, W2, Wk, Wq, Wv, Wo, arel, mrel,
                                      bk, bq, bv, bo, b1, b2, wf16, bpack, bsum, bzero);
  add3_kernel<<<(int)((nbThreads + 255) / 256), 256, 0, stream>>>(zL, zH, xemb, h, h16, nbThreads);

  // ---------------- GAT layers ----------------
  for (int layer = 0; layer < 2; ++layer) {
    long mcount = (long)NT * N_NODES * NH;
    fill_u32<<<(int)((mcount + 255) / 256), 256, 0, stream>>>(mbuf, 0x007FFFFFu, mcount);
    (void)hipMemsetAsync(sbuf, 0, (size_t)mcount * 4, stream);
    (void)hipMemsetAsync(agg, 0, NB * 4, stream);

    gemm128<false><<<dim3(gemmGx, NT), 256, 0, stream>>>(
        h16, wf16 + (size_t)layer * 4 * 16384, xw, bzero, 0L, N_NODES, (long)NB);

    const float* as_ = layer ? as2 : as1;
    const float* ad_ = layer ? ad2 : ad1;
    long ccount = (long)NT * N_NODES * NH;
    gat_coeff<<<(int)((ccount + 255) / 256), 256, 0, stream>>>(xw, as_, ad_, al, ar);

    long ecount = (long)NT * EPT;
    gat_edge_max<<<(int)((ecount + 255) / 256), 256, 0, stream>>>(al, ar, eip, mbuf);
    gat_edge_sum<<<(int)((ecount + 255) / 256), 256, 0, stream>>>(al, ar, eip, mbuf, sbuf);
    long acount = ecount * 32;
    gat_edge_agg<<<(int)((acount + 255) / 256), 256, 0, stream>>>(xw, al, ar, eip, mbuf, sbuf, agg);

    long rthreads = (long)N_NODES * 32;
    post_norm_gat<<<(int)((rthreads + 255) / 256), 256, 0, stream>>>(
        agg, bsum + layer * 128, layer ? g2 : g1, h, h16);
  }

  // ---------------- HGT layer ----------------
  {
    long mcount = (long)N_NODES * NH;
    fill_u32<<<(int)((mcount + 255) / 256), 256, 0, stream>>>(mbuf, 0x007FFFFFu, mcount);
    (void)hipMemsetAsync(sbuf, 0, (size_t)mcount * 4, stream);
    (void)hipMemsetAsync(agg, 0, NB * 4, stream);

    // k,q,v = h @ {Wk,Wq,Wv} + {bk,bq,bv}  (f16 outputs, slots 8..10)
    gemm128<true><<<dim3(gemmGx, 3), 256, 0, stream>>>(
        h16, wf16 + (size_t)8 * 16384, kqv16, bpack, 128L, N_NODES, (long)NB);
    // qa[t] = q @ blkdiag(arel[t])   (slots 12..15)
    gemm128<true><<<dim3(gemmGx, NT), 256, 0, stream>>>(
        kqv16 + NB, wf16 + (size_t)12 * 16384, qa16, bzero, 0L, N_NODES, (long)NB);
    // vm[t] = v @ blkdiag(mrel[t])   (slots 16..19)
    gemm128<true><<<dim3(gemmGx, NT), 256, 0, stream>>>(
        kqv16 + 2 * NB, wf16 + (size_t)16 * 16384, vm16, bzero, 0L, N_NODES, (long)NB);

    long tcount = (long)NT * NE * NH;
    hgt_att_kernel<<<(int)((tcount + 255) / 256), 256, 0, stream>>>(
        kqv16, qa16, prel, eip, att, mbuf);
    hgt_sum_kernel<<<(int)((tcount + 255) / 256), 256, 0, stream>>>(att, eip, mbuf, sbuf);
    long acount = (long)NT * NE * 32;
    hgt_agg_kernel<<<(int)((acount + 255) / 256), 256, 0, stream>>>(
        att, vm16, eip, mbuf, sbuf, agg);

    gelu16_kernel<<<(int)((nbThreads + 255) / 256), 256, 0, stream>>>(agg, ge16, nbThreads);
    // O = gelu(agg) @ Wo + bo  (slot 11, bias row 3)
    gemm128<false><<<dim3(gemmGx, 1), 256, 0, stream>>>(
        ge16, wf16 + (size_t)11 * 16384, Obuf, bpack + 384, 0L, N_NODES, (long)NB);

    long rthreads = (long)N_NODES * 32;
    final_kernel<<<(int)((rthreads + 255) / 256), 256, 0, stream>>>(
        Obuf, h, g3, skip, (float*)d_out);
  }
}
